// GenericBlock_48945447305842
// MI455X (gfx1250) — matrix operand order repaired and checked by reading, not compiled, not run on hardware
//
#include <hip/hip_runtime.h>
#include <math.h>

typedef __attribute__((ext_vector_type(16))) _Float16 v16h;
typedef __attribute__((ext_vector_type(8)))  float    v8f;

#define NB 512   // points
#define CIO 64   // channels

// ---------------- Kernel 1a: gnorm_dot[b,j] = sum_c ||geom[b,j,:,c]|| * nw[c]
__global__ __launch_bounds__(256) void node_norm_kernel(
    const float* __restrict__ geom, const float* __restrict__ norm_w,
    float* __restrict__ gnd) {
  int idx = blockIdx.x * blockDim.x + threadIdx.x;   // b*N + j
  if (idx >= 2 * NB) return;
  const float* g = geom + (size_t)idx * 3 * CIO;     // [3][64]
  float dot = 0.f;
  #pragma unroll 4
  for (int c = 0; c < CIO; ++c) {
    float g0 = g[c], g1 = g[CIO + c], g2 = g[2 * CIO + c];
    dot += sqrtf(g0 * g0 + g1 * g1 + g2 * g2) * norm_w[c];
  }
  gnd[idx] = dot;
}

// ---------------- Kernel 1b: gmh[b][o][d][j] = f16( sum_c geom[b,j,d,c]*theta[c,o] )
// Stored j-contiguous so the WMMA B-fragment is two contiguous 16B loads.
__global__ __launch_bounds__(256) void gm_kernel(
    const float* __restrict__ geom, const float* __restrict__ theta,
    _Float16* __restrict__ gmh) {
  int idx = blockIdx.x * blockDim.x + threadIdx.x;   // rowd*64 + o
  if (idx >= 2 * NB * 3 * CIO) return;
  int o = idx & 63;
  int rowd = idx >> 6;            // (b*N + j)*3 + d
  int d = rowd % 3;
  int bj = rowd / 3;              // b*N + j
  const float* g = geom + (size_t)rowd * CIO;
  float acc = 0.f;
  #pragma unroll 8
  for (int c = 0; c < CIO; ++c) acc += g[c] * theta[c * CIO + o];  // theta coalesced over o=lane
  int b = bj >> 9, j = bj & 511;
  gmh[(((b * CIO + o) * 3 + d) * NB) + j] = (_Float16)acc;
}

// ---------------- Kernel 2: pair scalar s[b,i,j]
__global__ __launch_bounds__(256) void pair_scalar_kernel(
    const float* __restrict__ rel, const float* __restrict__ emb,
    const float* __restrict__ gnd, const float* __restrict__ norm_w,
    const float* __restrict__ norm_b, float* __restrict__ s) {
  int idx = blockIdx.x * blockDim.x + threadIdx.x;   // (b*N + i)*N + j
  if (idx >= 2 * NB * NB) return;
  int j = idx & 511;
  int b = idx >> 18;
  const float* rp = rel + (size_t)idx * 3;
  float r0 = rp[0], r1 = rp[1], r2 = rp[2];
  float nrm = sqrtf(r0 * r0 + r1 * r1 + r2 * r2);
  s[idx] = gnd[b * NB + j] + nrm * norm_w[CIO] + norm_b[0] + emb[idx];
}

// ---------------- Kernel 3: main contraction.
// Block = 8 waves; block covers (b, 16-row i-tile, 32 o's); wave covers 4 o's.
// Per j-tile of 32: build A = w[i,k] (f16) per o from s/adj rows, B = gmh slice,
// accumulate with v_wmma_f32_16x16x32_f16; rel-term folded in-register.
__global__ __launch_bounds__(256) void gnn_main_kernel(
    const float* __restrict__ s, const float* __restrict__ adj,
    const float* __restrict__ rel, const _Float16* __restrict__ gmh,
    const float* __restrict__ theta, const float* __restrict__ alpha,
    const float* __restrict__ beta, float* __restrict__ out) {
  int bid = blockIdx.x;
  int b = bid >> 6;            // 64 blocks per batch
  int r = bid & 63;
  int i0 = (r >> 1) * 16;
  int ochunk = r & 1;
  int wave = threadIdx.x >> 5;
  int lane = threadIdx.x & 31;
  int obase = ochunk * 32 + wave * 4;

  // ISA wave32 fragment roles for this lane
  int row   = lane & 15;            // A-matrix row (i within tile)
  int kbase = (lane < 16) ? 0 : 8;  // A: lanes 0-15 hold K {0..7,16..23}, 16-31 hold {8..15,24..31}
  int n     = lane & 15;            // B/D column (= d, padded to 16)
  int khalf = lane >> 4;            // B: lanes 0-15 hold K {0..7,16..23}, 16-31 hold {8..15,24..31}
  int nd    = n < 3 ? n : 2;        // clamp for safe address; value zeroed below

  float al[4], be[4], th64[4];
  #pragma unroll
  for (int t = 0; t < 4; ++t) {
    al[t]   = alpha[obase + t];
    be[t]   = beta[obase + t];
    th64[t] = theta[CIO * CIO + obase + t];   // theta[64][o]
  }

  v8f acc[4] = {};
  float racc[4][3] = {};

  const float* srow = s   + ((b * NB + i0 + row) * NB);
  const float* arow = adj + ((b * NB + i0 + row) * NB);
  const float* rrow = rel + (size_t)((b * NB + i0 + row) * NB) * 3;
  const v16h zf = {};

  for (int j0 = 0; j0 < NB; j0 += 32) {
    const float* sp = srow + j0 + kbase;
    const float* ap = arow + j0 + kbase;
    float4 sq0 = *(const float4*)(sp);
    float4 sq1 = *(const float4*)(sp + 4);
    float4 sq2 = *(const float4*)(sp + 16);
    float4 sq3 = *(const float4*)(sp + 20);
    float4 aq0 = *(const float4*)(ap);
    float4 aq1 = *(const float4*)(ap + 4);
    float4 aq2 = *(const float4*)(ap + 16);
    float4 aq3 = *(const float4*)(ap + 20);
    float sv[16] = {sq0.x, sq0.y, sq0.z, sq0.w, sq1.x, sq1.y, sq1.z, sq1.w,
                    sq2.x, sq2.y, sq2.z, sq2.w, sq3.x, sq3.y, sq3.z, sq3.w};
    float av[16] = {aq0.x, aq0.y, aq0.z, aq0.w, aq1.x, aq1.y, aq1.z, aq1.w,
                    aq2.x, aq2.y, aq2.z, aq2.w, aq3.x, aq3.y, aq3.z, aq3.w};
    float rv[16][3];
    #pragma unroll
    for (int e = 0; e < 16; ++e) {
      int joff = kbase + (e < 8 ? e : e + 8);     // A-layout K index of element e
      const float* rp = rrow + (size_t)(j0 + joff) * 3;
      rv[e][0] = rp[0]; rv[e][1] = rp[1]; rv[e][2] = rp[2];
    }

    #pragma unroll
    for (int t = 0; t < 4; ++t) {
      int o = obase + t;
      v16h afrag;
      #pragma unroll
      for (int e = 0; e < 16; ++e) {
        float w = fmaxf(al[t] * sv[e] + be[t], 0.0f) * av[e];
        afrag[e] = (_Float16)w;
        racc[t][0] += rv[e][0] * w;
        racc[t][1] += rv[e][1] * w;
        racc[t][2] += rv[e][2] * w;
      }
      // B[k,n] = gm[b, j0+k, d=n, o]; lane holds j's {8h..8h+7, 16+8h..16+8h+7}
      const _Float16* gp = gmh + (((b * CIO + o) * 3 + nd) * NB + j0 + 8 * khalf);
      v16h bfrag;
      #pragma unroll
      for (int e = 0; e < 16; ++e) bfrag[e] = gp[e < 8 ? e : e + 8];
      if (n >= 3) bfrag = zf;                     // zero padded columns
      acc[t] = __builtin_amdgcn_wmma_f32_16x16x32_f16(
          false, afrag, false, bfrag, (short)0, acc[t], false, false);
    }
  }

  // Epilogue: fold lane-halves of rel-term, relocate to D-fragment layout, store.
  #pragma unroll
  for (int t = 0; t < 4; ++t) {
    int o = obase + t;
    float r0 = racc[t][0] + __shfl_xor(racc[t][0], 16, 32);
    float r1 = racc[t][1] + __shfl_xor(racc[t][1], 16, 32);
    float r2 = racc[t][2] + __shfl_xor(racc[t][2], 16, 32);
    #pragma unroll
    for (int rr = 0; rr < 8; ++rr) {
      int m = rr + 8 * khalf;                 // D row for VGPR rr in this lane half
      float v0 = __shfl(r0, m, 32);
      float v1 = __shfl(r1, m, 32);
      float v2 = __shfl(r2, m, 32);
      float rvn = (n == 0) ? v0 : ((n == 1) ? v1 : v2);
      if (n < 3) {
        out[(((b * NB + i0 + m) * 3) + n) * CIO + o] = acc[t][rr] + th64[t] * rvn;
      }
    }
  }
}

extern "C" void kernel_launch(void* const* d_in, const int* in_sizes, int n_in,
                              void* d_out, int out_size, void* d_ws, size_t ws_size,
                              hipStream_t stream) {
  const float* geom   = (const float*)d_in[0];  // [2,512,3,64]
  const float* adj    = (const float*)d_in[1];  // [2,512,512,1,1]
  const float* rel    = (const float*)d_in[2];  // [2,512,512,3,1]
  const float* emb    = (const float*)d_in[3];  // [2,512,512,1]
  const float* theta  = (const float*)d_in[4];  // [65,64]
  const float* norm_w = (const float*)d_in[5];  // [65,1]
  const float* norm_b = (const float*)d_in[6];  // [1]
  const float* alpha  = (const float*)d_in[7];  // [1,1,64]
  const float* beta   = (const float*)d_in[8];  // [1,1,64]
  float* out = (float*)d_out;                   // [2,512,3,64]

  char* ws = (char*)d_ws;
  float*     gnd = (float*)ws;                          // 1024 f32  (4 KB)
  _Float16*  gmh = (_Float16*)(ws + 4096);              // 2*64*3*512 f16 (384 KB)
  float*     sbf = (float*)(ws + 4096 + 393216);        // 2*512*512 f32 (2 MB)

  node_norm_kernel  <<<4,    256, 0, stream>>>(geom, norm_w, gnd);
  gm_kernel         <<<768,  256, 0, stream>>>(geom, theta, gmh);
  pair_scalar_kernel<<<2048, 256, 0, stream>>>(rel, emb, gnd, norm_w, norm_b, sbf);
  gnn_main_kernel   <<<128,  256, 0, stream>>>(sbf, adj, rel, gmh, theta, alpha, beta, out);
}
